// HGraphSAGE_29437705847373
// MI455X (gfx1250) — compile-verified, matrix operand order checked by reading, not run on hardware
//
#include <hip/hip_runtime.h>
#include <hip/hip_bf16.h>
#include <math.h>

// ---------------------------------------------------------------------------
// HAN / hetero-GAT forward for MI455X (gfx1250, wave32, WMMA).
// GEMMs use v_wmma_f32_16x16x32_f16 with pre-swizzled B fragments (global)
// and vector (b128) LDS loads for A fragments.
// ---------------------------------------------------------------------------

#define Nn 100000
#define Rr 3
#define Ee 320000
// HID = 128, H=4, D=32, HD=128

typedef __attribute__((ext_vector_type(16))) _Float16 v16h;
typedef __attribute__((ext_vector_type(8)))  _Float16 v8h;
typedef __attribute__((ext_vector_type(4)))  _Float16 v4h;
typedef __attribute__((ext_vector_type(8)))  float    v8f;

// Fragment element mapping for 16-bit WMMA operands:
// element i (0..16383) of swizzled B  ->  (k, n) of the 128x128 matrix.
// frag = kt*8+nt; lane 0-15 holds n=n0..n0+15 with K={kb..kb+7, kb+16..kb+23},
// lanes 16-31 same n with K shifted by 8 (kb = kt*32 + 8).
__device__ __forceinline__ void frag_kn(int i, int& k, int& n) {
  int j    = i & 15;
  int lane = (i >> 4) & 31;
  int frag = i >> 9;          // 0..31
  int nt = frag & 7, kt = frag >> 3;
  n = nt * 16 + (lane & 15);
  int kb = kt * 32 + ((lane & 16) ? 8 : 0);
  k = (j < 8) ? (kb + j) : (kb + 8 + j);   // kb+16+(j-8)
}

// Shared GEMM core: acc += A[row0..row0+127, 0:128] * B  (B pre-swizzled).
// 256 threads = 8 waves; wave w owns rows w*16..w*16+15 of the block tile.
__device__ __forceinline__ void gemm_core(const float* Ag, const _Float16* Bsw,
                                          int row0, int nrows,
                                          _Float16 As[][136], v8f acc[8]) {
  const int tid  = threadIdx.x;
  const int lane = tid & 31;
  const int m0   = (tid >> 5) * 16;
  // Stage full A tile (128x128), f32 -> f16, vector stores (ds_store_b64).
#pragma unroll
  for (int i = 0; i < 16; ++i) {
    int linear = (i * 256 + tid) * 4;
    int rr = linear >> 7, cc = linear & 127;
    int grow = row0 + rr;
    float4 v = make_float4(0.f, 0.f, 0.f, 0.f);
    if (grow < nrows) v = *(const float4*)&Ag[(size_t)grow * 128 + cc];
    v4h hv = { (_Float16)v.x, (_Float16)v.y, (_Float16)v.z, (_Float16)v.w };
    *(v4h*)&As[rr][cc] = hv;
  }
  __syncthreads();
  const int mrow = m0 + (lane & 15);
  const int klo  = (lane & 16) ? 8 : 0;
#pragma unroll
  for (int kt = 0; kt < 4; ++kt) {
    int kb = kt * 32 + klo;
    // A fragment: two aligned 16B runs -> 2x ds_load_b128.
    v8h alo = *(const v8h*)&As[mrow][kb];
    v8h ahi = *(const v8h*)&As[mrow][kb + 16];
    v16h af = __builtin_shufflevector(alo, ahi,
        0, 1, 2, 3, 4, 5, 6, 7, 8, 9, 10, 11, 12, 13, 14, 15);
    const _Float16* bbase = Bsw + (size_t)kt * 8 * 512;  // 512 halves / fragment
#pragma unroll
    for (int nt = 0; nt < 8; ++nt) {
      const _Float16* bp = bbase + ((size_t)nt * 32 + lane) * 16;
      v8h blo = *(const v8h*)bp;        // global_load_b128
      v8h bhi = *(const v8h*)(bp + 8);  // global_load_b128
      v16h bf = __builtin_shufflevector(blo, bhi,
          0, 1, 2, 3, 4, 5, 6, 7, 8, 9, 10, 11, 12, 13, 14, 15);
      acc[nt] = __builtin_amdgcn_wmma_f32_16x16x32_f16(
          false, af, false, bf, (short)0, acc[nt], false, false);
    }
  }
}

// ------------------------- weight preparation ------------------------------
// Folds: Wc_src[r] = Wt_src[r] @ Wg[r] (stored f16, fragment-swizzled),
//        bc = bt_src[r] @ Wg[r].
// Attention folds via Gfold[k][h] = sum_d Wg[k][h*32+d]*attn[h][d]:
//   Wel/Wer[f][h] = sum_k WT[f][k]*Gfold[k][h],  bel/ber from the bias.
__global__ void k_prep(const float* Wt_dst, const float* bt_dst,
                       const float* Wt_src, const float* bt_src,
                       const float* Wg, const float* attn_l, const float* attn_r,
                       _Float16* WcSw, float* bcSrc,
                       float* Wel, float* bel, float* Wer, float* ber) {
  const int r   = blockIdx.x;
  const int sel = blockIdx.y;  // 0 = src path (el, hs), 1 = dst path (er)
  const int tid = threadIdx.x;
  const float* A    = sel ? Wt_dst : (Wt_src + (size_t)r * 128 * 128);
  const float* bvec = sel ? bt_dst : (bt_src + r * 128);
  const float* G    = Wg + (size_t)r * 128 * 128;
  const float* att  = (sel ? attn_r : attn_l) + r * 128;  // [H=4][D=32]
  float* Wfold = (sel ? Wer : Wel) + r * 128 * 4;
  float* bfold = (sel ? ber : bel) + r * 4;
  __shared__ float sG4[128][4];
  for (int i = tid; i < 128 * 4; i += 256) {
    int k = i >> 2, h = i & 3;
    float s = 0.f;
    for (int d = 0; d < 32; ++d) s += G[k * 128 + h * 32 + d] * att[h * 32 + d];
    sG4[k][h] = s;
  }
  __syncthreads();
  for (int i = tid; i < 128 * 4; i += 256) {
    int f = i >> 2, h = i & 3;
    float s = 0.f;
    for (int k = 0; k < 128; ++k) s += A[f * 128 + k] * sG4[k][h];
    Wfold[i] = s;
  }
  if (tid < 4) {
    float s = 0.f;
    for (int k = 0; k < 128; ++k) s += bvec[k] * sG4[k][tid];
    bfold[tid] = s;
  }
  if (sel == 0) {
    // Wc in WMMA-fragment order, f16.
    _Float16* WcH = WcSw + (size_t)r * 128 * 128;
    for (int i = tid; i < 128 * 128; i += 256) {
      int k, n;
      frag_kn(i, k, n);
      float s = 0.f;
      for (int m = 0; m < 128; ++m) s += A[k * 128 + m] * G[m * 128 + n];
      WcH[i] = (_Float16)s;
    }
    if (tid < 128) {
      float s = 0.f;
      for (int k = 0; k < 128; ++k) s += bvec[k] * G[k * 128 + tid];
      bcSrc[r * 128 + tid] = s;
    }
  }
}

// W1 -> f16, fragment-swizzled.
__global__ void k_swzW1(const float* W1, _Float16* W1sw) {
  int i = blockIdx.x * 256 + threadIdx.x;
  if (i >= 128 * 128) return;
  int k, n;
  frag_kn(i, k, n);
  W1sw[i] = (_Float16)W1[k * 128 + n];
}

// ------------------------- hs projection GEMM ------------------------------
// hs[r] = src_feats[r] @ Wc_src[r] + bc[r]   (stores f32 [Rr,Nn,128])
__global__ void __launch_bounds__(256)
k_gemm_hs(const float* A, const _Float16* Bsw, const float* bias, float* out) {
  __shared__ __align__(16) _Float16 As[128][136];
  const int r    = blockIdx.y;
  const int row0 = blockIdx.x * 128;
  const float*    Ag = A   + (size_t)r * Nn * 128;
  const _Float16* Bg = Bsw + (size_t)r * 128 * 128;
  v8f acc[8] = {};
  gemm_core(Ag, Bg, row0, Nn, As, acc);
  const int lane = threadIdx.x & 31;
  const int m0   = (threadIdx.x >> 5) * 16;
  const int ml   = (lane & 16) ? 8 : 0;  // C/D layout: lanes 16-31 hold M=8..15
  float* Og = out + (size_t)r * Nn * 128;
  const float* bb = bias + r * 128;
#pragma unroll
  for (int nt = 0; nt < 8; ++nt) {
    int col = nt * 16 + (lane & 15);
    float bv = bb[col];
#pragma unroll
    for (int q = 0; q < 8; ++q) {
      int grow = row0 + m0 + ml + q;
      if (grow < Nn) Og[(size_t)grow * 128 + col] = acc[nt][q] + bv;
    }
  }
}

// ------------------------- attention logits --------------------------------
// el[r][n][h] = src_feats[r][n] . Wel[r][:,h] + bel ; er from dst_feat / Wer.
__global__ void __launch_bounds__(256)
k_el_er(const float* dst_feat, const float* src_feats,
        const float* Wel, const float* bel, const float* Wer, const float* ber,
        float* el, float* er) {
  __shared__ float sWel[Rr * 128 * 4];
  __shared__ float sWer[Rr * 128 * 4];
  __shared__ float sbel[Rr * 4];
  __shared__ float sber[Rr * 4];
  const int tid = threadIdx.x;
  for (int i = tid; i < Rr * 128 * 4; i += 256) { sWel[i] = Wel[i]; sWer[i] = Wer[i]; }
  if (tid < Rr * 4) { sbel[tid] = bel[tid]; sber[tid] = ber[tid]; }
  __syncthreads();
  const int n = blockIdx.x * 256 + tid;
  if (n >= Nn) return;
  float ac[Rr][4];
#pragma unroll
  for (int r = 0; r < Rr; ++r)
#pragma unroll
    for (int h = 0; h < 4; ++h) ac[r][h] = 0.f;
  const float* xp = &dst_feat[(size_t)n * 128];
#pragma unroll 4
  for (int f4 = 0; f4 < 32; ++f4) {
    float4 x = *(const float4*)&xp[f4 * 4];
#pragma unroll
    for (int r = 0; r < Rr; ++r) {
      const float* wp = &sWer[(r * 128 + f4 * 4) * 4];
#pragma unroll
      for (int h = 0; h < 4; ++h)
        ac[r][h] += x.x * wp[h] + x.y * wp[4 + h] + x.z * wp[8 + h] + x.w * wp[12 + h];
    }
  }
#pragma unroll
  for (int r = 0; r < Rr; ++r) {
    float4 o;
    o.x = ac[r][0] + sber[r * 4 + 0];
    o.y = ac[r][1] + sber[r * 4 + 1];
    o.z = ac[r][2] + sber[r * 4 + 2];
    o.w = ac[r][3] + sber[r * 4 + 3];
    *(float4*)&er[((size_t)r * Nn + n) * 4] = o;
  }
#pragma unroll 1
  for (int r = 0; r < Rr; ++r) {
    float a0 = 0.f, a1 = 0.f, a2 = 0.f, a3 = 0.f;
    const float* sp = &src_feats[((size_t)r * Nn + n) * 128];
#pragma unroll 4
    for (int f4 = 0; f4 < 32; ++f4) {
      float4 x = *(const float4*)&sp[f4 * 4];
      const float* wp = &sWel[(r * 128 + f4 * 4) * 4];
      a0 += x.x * wp[0] + x.y * wp[4]  + x.z * wp[8]  + x.w * wp[12];
      a1 += x.x * wp[1] + x.y * wp[5]  + x.z * wp[9]  + x.w * wp[13];
      a2 += x.x * wp[2] + x.y * wp[6]  + x.z * wp[10] + x.w * wp[14];
      a3 += x.x * wp[3] + x.y * wp[7]  + x.z * wp[11] + x.w * wp[15];
    }
    float4 o;
    o.x = a0 + sbel[r * 4 + 0];
    o.y = a1 + sbel[r * 4 + 1];
    o.z = a2 + sbel[r * 4 + 2];
    o.w = a3 + sbel[r * 4 + 3];
    *(float4*)&el[((size_t)r * Nn + n) * 4] = o;
  }
}

// ------------------------- edge phase --------------------------------------
// Order-preserving float<->uint encoding for atomic segment-max.
__device__ __forceinline__ unsigned fenc(float f) {
  unsigned b = __float_as_uint(f);
  return (b & 0x80000000u) ? ~b : (b | 0x80000000u);
}
__device__ __forceinline__ float fdec(unsigned u) {
  unsigned b = (u & 0x80000000u) ? (u & 0x7fffffffu) : ~u;
  return __uint_as_float(b);
}
#define ENC_NEG_INF 0x007FFFFFu

__global__ void k_edge_max(const int* src_idx, const int* dst_idx,
                           const float* el, const float* er,
                           float* ebuf, unsigned* menc) {
  int g = blockIdx.x * 256 + threadIdx.x;
  if (g >= Rr * Ee) return;
  int r = g / Ee;
  int s = src_idx[g], d = dst_idx[g];
  float4 l4 = *(const float4*)&el[((size_t)r * Nn + s) * 4];
  float4 r4 = *(const float4*)&er[((size_t)r * Nn + d) * 4];
  float4 e;
  float v;
  v = l4.x + r4.x; e.x = v > 0.f ? v : 0.2f * v;
  v = l4.y + r4.y; e.y = v > 0.f ? v : 0.2f * v;
  v = l4.z + r4.z; e.z = v > 0.f ? v : 0.2f * v;
  v = l4.w + r4.w; e.w = v > 0.f ? v : 0.2f * v;
  *(float4*)&ebuf[(size_t)g * 4] = e;
  unsigned* mp = &menc[((size_t)r * Nn + d) * 4];
  atomicMax(&mp[0], fenc(e.x));
  atomicMax(&mp[1], fenc(e.y));
  atomicMax(&mp[2], fenc(e.z));
  atomicMax(&mp[3], fenc(e.w));
}

__global__ void k_edge_exp(const int* dst_idx, const unsigned* menc,
                           float* ebuf, float* ssum) {
  int g = blockIdx.x * 256 + threadIdx.x;
  if (g >= Rr * Ee) return;
  int r = g / Ee;
  int d = dst_idx[g];
  float4 e = *(const float4*)&ebuf[(size_t)g * 4];
  const unsigned* mp = &menc[((size_t)r * Nn + d) * 4];
  float4 ex;
  ex.x = __expf(e.x - fdec(mp[0]));
  ex.y = __expf(e.y - fdec(mp[1]));
  ex.z = __expf(e.z - fdec(mp[2]));
  ex.w = __expf(e.w - fdec(mp[3]));
  *(float4*)&ebuf[(size_t)g * 4] = ex;
  float* sp = &ssum[((size_t)r * Nn + d) * 4];
  atomicAdd(&sp[0], ex.x);
  atomicAdd(&sp[1], ex.y);
  atomicAdd(&sp[2], ex.z);
  atomicAdd(&sp[3], ex.w);
}

// One wave32 per edge: each lane handles 4 consecutive features (128 total).
__global__ void __launch_bounds__(256)
k_edge_agg(const int* src_idx, const int* dst_idx,
           const float* ebuf, const float* ssum, const float* hs, float* rst) {
  int wid  = blockIdx.x * 8 + (threadIdx.x >> 5);
  int lane = threadIdx.x & 31;
  if (wid >= Rr * Ee) return;
  int r = wid / Ee;
  int s = src_idx[wid], d = dst_idx[wid];
  float4 ex = *(const float4*)&ebuf[(size_t)wid * 4];
  float4 ss = *(const float4*)&ssum[((size_t)r * Nn + d) * 4];
  float a0 = ex.x / (ss.x + 1e-9f);
  float a1 = ex.y / (ss.y + 1e-9f);
  float a2 = ex.z / (ss.z + 1e-9f);
  float a3 = ex.w / (ss.w + 1e-9f);
  int h = lane >> 3;  // head index = (lane*4)/32
  float aa = (h == 0) ? a0 : (h == 1) ? a1 : (h == 2) ? a2 : a3;
  const float4 hv = *(const float4*)&hs[((size_t)r * Nn + s) * 128 + lane * 4];
  float* rp = &rst[((size_t)r * Nn + d) * 128 + lane * 4];
  atomicAdd(&rp[0], aa * hv.x);
  atomicAdd(&rp[1], aa * hv.y);
  atomicAdd(&rp[2], aa * hv.z);
  atomicAdd(&rp[3], aa * hv.w);
}

__global__ void k_bias_elu(const float* bias_g, float* rst) {
  long long i = (long long)blockIdx.x * 256 + threadIdx.x;
  const long long total = (long long)Rr * Nn * 128;
  if (i >= total) return;
  int col = (int)(i & 127);
  int r   = (int)(i / ((long long)Nn * 128));
  float v = rst[i] + bias_g[r * 128 + col];
  rst[i] = v > 0.f ? v : (__expf(v) - 1.f);
}

// ------------------------- semantic attention ------------------------------
// wsum[r] += sum over tile rows of ( tanh(z@W1 + b1) . W2 )  (WMMA + epilogue)
__global__ void __launch_bounds__(256)
k_sem(const float* Z, const _Float16* W1sw, const float* b1, const float* W2,
      float* wsum) {
  __shared__ __align__(16) _Float16 As[128][136];
  const int r    = blockIdx.y;
  const int row0 = blockIdx.x * 128;
  const float* Ag = Z + (size_t)r * Nn * 128;
  v8f acc[8] = {};
  gemm_core(Ag, W1sw, row0, Nn, As, acc);
  const int lane = threadIdx.x & 31;
  const int m0   = (threadIdx.x >> 5) * 16;
  const int ml   = (lane & 16) ? 8 : 0;
  float pl = 0.f;
#pragma unroll
  for (int nt = 0; nt < 8; ++nt) {
    int col = nt * 16 + (lane & 15);
    float b1v = b1[col];
    float w2v = W2[col];
#pragma unroll
    for (int q = 0; q < 8; ++q) {
      int grow = row0 + m0 + ml + q;
      if (grow < Nn) pl += tanhf(acc[nt][q] + b1v) * w2v;
    }
  }
#pragma unroll
  for (int off = 16; off > 0; off >>= 1) pl += __shfl_xor(pl, off, 32);
  if (lane == 0) atomicAdd(&wsum[r], pl);
}

__global__ void k_soft(const float* wsum, float* a_ws, float* out_att) {
  if (threadIdx.x == 0 && blockIdx.x == 0) {
    float w0 = wsum[0] / (float)Nn;
    float w1 = wsum[1] / (float)Nn;
    float w2 = wsum[2] / (float)Nn;
    float mx = fmaxf(w0, fmaxf(w1, w2));
    float e0 = __expf(w0 - mx), e1 = __expf(w1 - mx), e2 = __expf(w2 - mx);
    float inv = 1.f / (e0 + e1 + e2);
    a_ws[0] = e0 * inv; a_ws[1] = e1 * inv; a_ws[2] = e2 * inv;
    out_att[0] = a_ws[0]; out_att[1] = a_ws[1]; out_att[2] = a_ws[2];
  }
}

__global__ void k_final(const float* z, const float* a_ws, float* out) {
  long long i4 = (long long)blockIdx.x * 256 + threadIdx.x;
  const long long total4 = (long long)Nn * 128 / 4;
  if (i4 >= total4) return;
  float a0 = a_ws[0], a1 = a_ws[1], a2 = a_ws[2];
  const float4* zp = (const float4*)z;
  float4 v0 = zp[i4];
  float4 v1 = zp[i4 + total4];
  float4 v2 = zp[i4 + 2 * total4];
  float4 o;
  o.x = a0 * v0.x + a1 * v1.x + a2 * v2.x;
  o.y = a0 * v0.y + a1 * v1.y + a2 * v2.y;
  o.z = a0 * v0.z + a1 * v1.z + a2 * v2.z;
  o.w = a0 * v0.w + a1 * v1.w + a2 * v2.w;
  ((float4*)out)[i4] = o;
}

// ------------------------- fills (graph-capture safe) ----------------------
__global__ void k_fill_f32(float* p, float v, long long n) {
  long long i  = (long long)blockIdx.x * blockDim.x + threadIdx.x;
  long long st = (long long)gridDim.x * blockDim.x;
  for (; i < n; i += st) p[i] = v;
}
__global__ void k_fill_u32(unsigned* p, unsigned v, long long n) {
  long long i  = (long long)blockIdx.x * blockDim.x + threadIdx.x;
  long long st = (long long)gridDim.x * blockDim.x;
  for (; i < n; i += st) p[i] = v;
}

// ---------------------------------------------------------------------------
extern "C" void kernel_launch(void* const* d_in, const int* in_sizes, int n_in,
                              void* d_out, int out_size, void* d_ws, size_t ws_size,
                              hipStream_t stream) {
  (void)in_sizes; (void)n_in; (void)out_size; (void)ws_size;
  const float* dst_feat  = (const float*)d_in[0];
  const float* src_feats = (const float*)d_in[1];
  const int*   src_idx   = (const int*)d_in[2];
  const int*   dst_idx   = (const int*)d_in[3];
  const float* Wt_dst    = (const float*)d_in[4];
  const float* bt_dst    = (const float*)d_in[5];
  const float* Wt_src    = (const float*)d_in[6];
  const float* bt_src    = (const float*)d_in[7];
  const float* Wg        = (const float*)d_in[8];
  const float* attn_l    = (const float*)d_in[9];
  const float* attn_r    = (const float*)d_in[10];
  const float* bias_g    = (const float*)d_in[11];
  const float* W1        = (const float*)d_in[12];
  const float* b1        = (const float*)d_in[13];
  const float* W2        = (const float*)d_in[14];
  float* out = (float*)d_out;
  char*  ws  = (char*)d_ws;

  // ---- workspace layout (256B aligned) ----
  size_t off = 0;
  auto take = [&](size_t bytes) { size_t o = off; off += (bytes + 255) & ~(size_t)255; return o; };
  size_t oWcSw  = take((size_t)Rr * 128 * 128 * sizeof(_Float16));
  size_t oBcSrc = take((size_t)Rr * 128 * sizeof(float));
  size_t oWel   = take((size_t)Rr * 128 * 4 * sizeof(float));
  size_t oBel   = take(256);
  size_t oWer   = take((size_t)Rr * 128 * 4 * sizeof(float));
  size_t oBer   = take(256);
  size_t oW1Sw  = take((size_t)128 * 128 * sizeof(_Float16));
  size_t oEl    = take((size_t)Rr * Nn * 4 * sizeof(float));
  size_t oEr    = take((size_t)Rr * Nn * 4 * sizeof(float));
  size_t oHs    = take((size_t)Rr * Nn * 128 * sizeof(float));
  size_t oEbuf  = take((size_t)Rr * Ee * 4 * sizeof(float));
  size_t oMenc  = take((size_t)Rr * Nn * 4 * sizeof(unsigned));
  size_t oSsum  = take((size_t)Rr * Nn * 4 * sizeof(float));
  size_t oRst   = take((size_t)Rr * Nn * 128 * sizeof(float));
  size_t oWsum  = take(256);
  size_t oAvec  = take(256);

  _Float16* WcSw  = (_Float16*)(ws + oWcSw);
  float*    bcSrc = (float*)(ws + oBcSrc);
  float*    Welp  = (float*)(ws + oWel);
  float*    belp  = (float*)(ws + oBel);
  float*    Werp  = (float*)(ws + oWer);
  float*    berp  = (float*)(ws + oBer);
  _Float16* W1sw  = (_Float16*)(ws + oW1Sw);
  float*    elp   = (float*)(ws + oEl);
  float*    erp   = (float*)(ws + oEr);
  float*    hsp   = (float*)(ws + oHs);
  float*    ebufp = (float*)(ws + oEbuf);
  unsigned* mencp = (unsigned*)(ws + oMenc);
  float*    ssump = (float*)(ws + oSsum);
  float*    rstp  = (float*)(ws + oRst);
  float*    wsump = (float*)(ws + oWsum);
  float*    aptr  = (float*)(ws + oAvec);

  // ---- init accumulators ----
  k_fill_f32<<<2048, 256, 0, stream>>>(ssump, 0.f, (long long)Rr * Nn * 4);
  k_fill_f32<<<4096, 256, 0, stream>>>(rstp, 0.f, (long long)Rr * Nn * 128);
  k_fill_f32<<<1, 64, 0, stream>>>(wsump, 0.f, 8);
  k_fill_u32<<<2048, 256, 0, stream>>>(mencp, ENC_NEG_INF, (long long)Rr * Nn * 4);

  // ---- weight prep ----
  k_prep<<<dim3(Rr, 2), 256, 0, stream>>>(Wt_dst, bt_dst, Wt_src, bt_src, Wg,
                                          attn_l, attn_r, WcSw, bcSrc,
                                          Welp, belp, Werp, berp);
  k_swzW1<<<64, 256, 0, stream>>>(W1, W1sw);

  // ---- hs projection (WMMA) ----
  const int rowBlocks = (Nn + 127) / 128;
  k_gemm_hs<<<dim3(rowBlocks, Rr), 256, 0, stream>>>(src_feats, WcSw, bcSrc, hsp);

  // ---- attention logits ----
  k_el_er<<<(Nn + 255) / 256, 256, 0, stream>>>(dst_feat, src_feats,
                                                Welp, belp, Werp, berp, elp, erp);

  // ---- edge softmax + aggregation ----
  const int edgeBlocks = (Rr * Ee + 255) / 256;
  k_edge_max<<<edgeBlocks, 256, 0, stream>>>(src_idx, dst_idx, elp, erp, ebufp, mencp);
  k_edge_exp<<<edgeBlocks, 256, 0, stream>>>(dst_idx, mencp, ebufp, ssump);
  k_edge_agg<<<(Rr * Ee + 7) / 8, 256, 0, stream>>>(src_idx, dst_idx, ebufp, ssump,
                                                    hsp, rstp);
  k_bias_elu<<<(int)(((long long)Rr * Nn * 128 + 255) / 256), 256, 0, stream>>>(bias_g, rstp);

  // ---- semantic attention (WMMA) + combine ----
  k_sem<<<dim3(rowBlocks, Rr), 256, 0, stream>>>(rstp, W1sw, b1, W2, wsump);
  k_soft<<<1, 32, 0, stream>>>(wsump, aptr, out + (size_t)Nn * 128);
  k_final<<<(int)(((long long)Nn * 128 / 4 + 255) / 256), 256, 0, stream>>>(rstp, aptr, out);
}
